// NAMClassifier_62319975465245
// MI455X (gfx1250) — matrix operand order repaired and checked by reading, not compiled, not run on hardware
//
#include <hip/hip_runtime.h>

#define NFEAT 256
#define NH1   64
#define NH2   32

typedef _Float16 h2v  __attribute__((ext_vector_type(2)));
typedef __fp16   fp16v2 __attribute__((ext_vector_type(2)));
typedef _Float16 h8   __attribute__((ext_vector_type(8)));
typedef _Float16 v16h __attribute__((ext_vector_type(16)));
typedef float    v8f  __attribute__((ext_vector_type(8)));
typedef float    f4   __attribute__((ext_vector_type(4)));

// Per-feature staged parameters, f16-converted, with W2 pre-swizzled into the
// contiguous-k v_wmma_f32_16x16x32_f16 B-fragment order used here (matches A):
//   frag = 2*kc + nc  (kc = K/32 chunk, nc = N/16 half)
//   within frag: [lane][e] where lane = 16*(Klocal/16) + (N%16),
//                e in 0..15 maps to Klocal = 16*(lane/16) + e
// so each lane's fragment strip is 16 contiguous halfs (32B) in LDS.
struct __align__(16) Staged {
  _Float16 w2f[4 * 32 * 16];   // 4 KB
  _Float16 w1h[NH1];
  _Float16 b1h[NH1];
  float    b2s[NH2];
  float    w3s[NH2];
};

__device__ __forceinline__ h8 relu8(h8 t) {
#if __has_builtin(__builtin_elementwise_max)
  h8 z = {};
  return __builtin_elementwise_max(t, z);   // -> v_pk_max_num_f16
#else
  #pragma unroll
  for (int i = 0; i < 8; ++i) t[i] = t[i] > (_Float16)0 ? t[i] : (_Float16)0;
  return t;
#endif
}

__device__ __forceinline__ void stage_feature(
    Staged& s, int tid, int f,
    const float* __restrict__ W1, const float* __restrict__ b1,
    const float* __restrict__ W2, const float* __restrict__ b2,
    const float* __restrict__ W3)
{
  // ---- W2[f] : [64][32] fp32 -> f16 B-fragments ----
  // thread t owns K-pair (2m, 2m+1), N = nb..nb+7  => 8 packed half2 LDS stores
  const float* W2f = W2 + (size_t)f * (NH1 * NH2);
  const int m  = tid >> 2;            // 0..31
  const int nb = (tid & 3) << 3;      // 0,8,16,24
  const float* r0 = W2f + (2 * m) * NH2 + nb;
  const float* r1 = r0 + NH2;
  f4 r0a = *(const f4*)(r0);  f4 r0b = *(const f4*)(r0 + 4);
  f4 r1a = *(const f4*)(r1);  f4 r1b = *(const f4*)(r1 + 4);
  const int kc = m >> 4;
  const int hi = (m >> 3) & 1;
  const int p  = m & 7;               // element pair: e = 2p (K, K+1 share a dword)
  #pragma unroll
  for (int j = 0; j < 8; ++j) {
    float vlo = (j < 4) ? r0a[j] : r0b[j - 4];
    float vhi = (j < 4) ? r1a[j] : r1b[j - 4];
    int N    = nb + j;
    int nc   = N >> 4;
    int lane = (hi << 4) | (N & 15);
    int frag = (kc << 1) | nc;
    // one v_cvt_pk_rtz_f16_f32 per K-pair; store via __fp16 vec (same layout)
    fp16v2 pk = __builtin_amdgcn_cvt_pkrtz(vlo, vhi);
    *(fp16v2*)&s.w2f[frag * 512 + lane * 16 + 2 * p] = pk;
  }
  // ---- W1 / b1 -> f16 ----
  if (tid < NH1) s.w1h[tid]        = (_Float16)W1[(size_t)f * NH1 + tid];
  else           s.b1h[tid - NH1]  = (_Float16)b1[(size_t)f * NH1 + (tid - NH1)];
  // ---- b2 / W3 stay fp32 ----
  if (tid < NH2)             s.b2s[tid]        = b2[(size_t)f * NH2 + tid];
  else if (tid < 2 * NH2)    s.w3s[tid - NH2]  = W3[(size_t)f * NH2 + (tid - NH2)];
}

__global__ __launch_bounds__(128, 1) void nam_mlp_wmma(
    const float* __restrict__ x,  const float* __restrict__ W1,
    const float* __restrict__ b1, const float* __restrict__ W2,
    const float* __restrict__ b2, const float* __restrict__ W3,
    const float* __restrict__ b3, const float* __restrict__ bias,
    float* __restrict__ out)
{
  __shared__ Staged st[2];
  const int tid   = threadIdx.x;
  const int lane  = tid & 31;
  const int wave  = tid >> 5;
  const int laneN = lane & 15;      // N column (C/D) == M row (A)
  const int hiH   = lane >> 4;      // lane half
  const int rowBase = blockIdx.x * 64 + wave * 16;
  const float* xrow = x + (size_t)(rowBase + laneN) * NFEAT;

  float acc[8];
  #pragma unroll
  for (int r = 0; r < 8; ++r) acc[r] = 0.f;

  stage_feature(st[0], tid, 0, W1, b1, W2, b2, W3);
  __syncthreads();

  for (int f = 0; f < NFEAT; ++f) {
    if (f + 1 < NFEAT)
      stage_feature(st[(f + 1) & 1], tid, f + 1, W1, b1, W2, b2, W3);

    const Staged& s = st[f & 1];

    // ---- build A = relu(x * W1 + b1), 16x64, as two 16x32 f16 fragments ----
    float xv = xrow[f];
    _Float16 xh = (_Float16)xv;
    h8 x8 = { xh, xh, xh, xh, xh, xh, xh, xh };
    const h8* w1p = (const h8*)s.w1h;   // 8-half (16B) strips
    const h8* b1p = (const h8*)s.b1h;
    h8 t00 = relu8(x8 * w1p[0 + 2 * hiH] + b1p[0 + 2 * hiH]);  // K = 16*hi + e
    h8 t01 = relu8(x8 * w1p[1 + 2 * hiH] + b1p[1 + 2 * hiH]);  // K = 16*hi + 8 + e
    h8 t10 = relu8(x8 * w1p[4 + 2 * hiH] + b1p[4 + 2 * hiH]);  // K = 32 + ...
    h8 t11 = relu8(x8 * w1p[5 + 2 * hiH] + b1p[5 + 2 * hiH]);  // K = 40 + 16*hi + e
    v16h a0 = __builtin_shufflevector(t00, t01, 0,1,2,3,4,5,6,7,8,9,10,11,12,13,14,15);
    v16h a1 = __builtin_shufflevector(t10, t11, 0,1,2,3,4,5,6,7,8,9,10,11,12,13,14,15);

    // ---- load W2 B-fragments: 32B contiguous per lane per fragment ----
    const h8* w2p = (const h8*)s.w2f;
    const int bofs = lane * 2;
    h8 q0 = w2p[bofs + 0 * 64], q1 = w2p[bofs + 0 * 64 + 1];
    h8 q2 = w2p[bofs + 1 * 64], q3 = w2p[bofs + 1 * 64 + 1];
    h8 q4 = w2p[bofs + 2 * 64], q5 = w2p[bofs + 2 * 64 + 1];
    h8 q6 = w2p[bofs + 3 * 64], q7 = w2p[bofs + 3 * 64 + 1];
    v16h b00 = __builtin_shufflevector(q0, q1, 0,1,2,3,4,5,6,7,8,9,10,11,12,13,14,15);
    v16h b01 = __builtin_shufflevector(q2, q3, 0,1,2,3,4,5,6,7,8,9,10,11,12,13,14,15);
    v16h b10 = __builtin_shufflevector(q4, q5, 0,1,2,3,4,5,6,7,8,9,10,11,12,13,14,15);
    v16h b11 = __builtin_shufflevector(q6, q7, 0,1,2,3,4,5,6,7,8,9,10,11,12,13,14,15);

    float b2lo = s.b2s[laneN],      b2hi = s.b2s[16 + laneN];
    float w3lo = s.w3s[laneN],      w3hi = s.w3s[16 + laneN];

    // ---- h2pre = A x W2 ; zero C folds to inline-0 SRC2 (no init movs) ----
    v8f c0 = {}, c1 = {};
    c0 = __builtin_amdgcn_wmma_f32_16x16x32_f16(false, a0, false, b00, (short)0, c0, false, false);
    c0 = __builtin_amdgcn_wmma_f32_16x16x32_f16(false, a1, false, b10, (short)0, c0, false, false);
    c1 = __builtin_amdgcn_wmma_f32_16x16x32_f16(false, a0, false, b01, (short)0, c1, false, false);
    c1 = __builtin_amdgcn_wmma_f32_16x16x32_f16(false, a1, false, b11, (short)0, c1, false, false);

    // ---- +b2, relu, dot with W3; add feeds maxnum so no canonicalize max ----
    #pragma unroll
    for (int r = 0; r < 8; ++r) {
      float h0  = __builtin_fmaxf(c0[r] + b2lo, 0.f);
      float h1v = __builtin_fmaxf(c1[r] + b2hi, 0.f);
      acc[r] = __builtin_fmaf(h0, w3lo, __builtin_fmaf(h1v, w3hi, acc[r]));
    }

    __syncthreads();   // single barrier/feature: double-buffered staging
  }

  // ---- sum of b3 over features (once, coalesced) ----
  float pb3 = b3[lane];
  #pragma unroll
  for (int k = 1; k < NFEAT / 32; ++k) pb3 += b3[lane + 32 * k];
  #pragma unroll
  for (int msk = 1; msk < 32; msk <<= 1) pb3 += __shfl_xor(pb3, msk, 32);

  // ---- reduce over the 16 lanes sharing each row-half; write rows ----
  #pragma unroll
  for (int r = 0; r < 8; ++r) {
    float v = acc[r];
    v += __shfl_xor(v, 1, 32);
    v += __shfl_xor(v, 2, 32);
    v += __shfl_xor(v, 4, 32);
    v += __shfl_xor(v, 8, 32);
    acc[r] = v;
  }
  if (laneN == 0) {
    float tail = pb3 + bias[0];
    #pragma unroll
    for (int r = 0; r < 8; ++r)
      out[rowBase + (hiH << 3) + r] = acc[r] + tail;   // row M = r + 8*hi
  }
}

extern "C" void kernel_launch(void* const* d_in, const int* in_sizes, int n_in,
                              void* d_out, int out_size, void* d_ws, size_t ws_size,
                              hipStream_t stream) {
  (void)n_in; (void)out_size; (void)d_ws; (void)ws_size;
  const float* x    = (const float*)d_in[0];
  const float* W1   = (const float*)d_in[1];
  const float* b1   = (const float*)d_in[2];
  const float* W2   = (const float*)d_in[3];
  const float* b2   = (const float*)d_in[4];
  const float* W3   = (const float*)d_in[5];
  const float* b3   = (const float*)d_in[6];
  const float* bias = (const float*)d_in[7];
  float* out = (float*)d_out;

  const int B    = in_sizes[0] / NFEAT;   // 16384
  const int grid = B / 64;                // 64 rows per 128-thread block
  nam_mlp_wmma<<<grid, 128, 0, stream>>>(x, W1, b1, W2, b2, W3, b3, bias, out);
}
